// MultiAttention_58514634441352
// MI455X (gfx1250) — matrix operand order repaired and checked by reading, not compiled, not run on hardware
//
#include <hip/hip_runtime.h>
#include <math.h>

#define Hdim  1024
#define Tdim  1024
#define Sdim  1024
#define NHEAD 16
#define DHEAD 64

typedef __attribute__((ext_vector_type(16))) _Float16 v16h;
typedef __attribute__((ext_vector_type(8)))  _Float16 v8h;
typedef __attribute__((ext_vector_type(8)))  float    v8f;

union V16 { v16h v; v8h h[2]; };

// ---------------------------------------------------------------------------
// Kernel 1: C = relu(X @ W^T + bias), output f16 into head-split layouts.
// mode 0/1: out[(b*16+h)*1024 + t][d]   (Q / K)
// mode 2  : out[(b*16+h)*64   + d][s]   (V transposed, for PV B-fragments)
// Block tile 128(M) x 128(N), BK=32, 8 waves as 4(M) x 2(N); each wave owns a
// 32x64 sub-tile = 2x4 WMMA tiles -> 8 back-to-back WMMAs per K-step.
// ---------------------------------------------------------------------------
__global__ __launch_bounds__(256)
void qkv_gemm_relu(const float* __restrict__ X, const float* __restrict__ W,
                   const float* __restrict__ bias, _Float16* __restrict__ out,
                   int mode) {
  __shared__ __attribute__((aligned(16))) _Float16 At[128][32];
  __shared__ __attribute__((aligned(16))) _Float16 Bt[128][32];
  const int tid  = threadIdx.x;
  const int wave = tid >> 5;
  const int lane = tid & 31;
  const int hl   = lane >> 4;   // which 16-lane half
  const int l16  = lane & 15;
  const int M0 = blockIdx.x * 128;
  const int N0 = blockIdx.y * 128;
  const int wm = wave >> 1;     // 0..3  (M: 2 tiles each)
  const int wn = wave & 1;      // 0..1  (N: 4 tiles each)

  const v8f vzero = {0.f,0.f,0.f,0.f,0.f,0.f,0.f,0.f};
  v8f acc[2][4];
#pragma unroll
  for (int mt = 0; mt < 2; ++mt)
#pragma unroll
    for (int nt = 0; nt < 4; ++nt) acc[mt][nt] = vzero;

  const int sr = tid >> 1;           // 0..127 staging row
  const int sc = (tid & 1) * 16;     // 0 / 16 staging col

  for (int k0 = 0; k0 < Hdim; k0 += 32) {
    { // stage A tile (128x32), f32 -> f16
      const float* src = X + (size_t)(M0 + sr) * Hdim + k0 + sc;
#pragma unroll
      for (int i = 0; i < 16; ++i) At[sr][sc + i] = (_Float16)src[i];
    }
    { // stage B tile (128x32): Bt[n][k] = W[N0+n][k0+k]
      const float* src = W + (size_t)(N0 + sr) * Hdim + k0 + sc;
#pragma unroll
      for (int i = 0; i < 16; ++i) Bt[sr][sc + i] = (_Float16)src[i];
    }
    __syncthreads();

    // preload ALL fragments, then fire 8 WMMAs back-to-back
    V16 a[2];
#pragma unroll
    for (int mt = 0; mt < 2; ++mt) {
      const _Float16* ap = &At[(wm * 2 + mt) * 16 + l16][0];
      a[mt].h[0] = *(const v8h*)(ap + 16 * hl);       // K = j + 16*hl
      a[mt].h[1] = *(const v8h*)(ap + 16 * hl + 8);   // K = 8 + j + 16*hl
    }
    V16 bm[4];
#pragma unroll
    for (int nt = 0; nt < 4; ++nt) {
      const _Float16* bp = &Bt[(wn * 4 + nt) * 16 + l16][16 * hl]; // K = j + 16*hl
      bm[nt].h[0] = *(const v8h*)bp;
      bm[nt].h[1] = *(const v8h*)(bp + 8);
    }
#pragma unroll
    for (int mt = 0; mt < 2; ++mt)
#pragma unroll
      for (int nt = 0; nt < 4; ++nt)
        acc[mt][nt] = __builtin_amdgcn_wmma_f32_16x16x32_f16(
            false, a[mt].v, false, bm[nt].v, (short)0, acc[mt][nt], false, false);
    __syncthreads();
  }

  // epilogue: bias + relu + f16 store into head-split layout
#pragma unroll
  for (int nt = 0; nt < 4; ++nt) {
    const int n  = N0 + (wn * 4 + nt) * 16 + l16;
    const float bn = bias[n];
    const int hh = n >> 6;
    const int d  = n & 63;
#pragma unroll
    for (int mt = 0; mt < 2; ++mt) {
#pragma unroll
      for (int r = 0; r < 8; ++r) {
        const int m  = M0 + (wm * 2 + mt) * 16 + r + 8 * hl;  // = b*1024 + t
        const int bb = m >> 10;
        const int t  = m & 1023;
        float v = acc[mt][nt][r] + bn;
        v = v > 0.f ? v : 0.f;
        size_t idx;
        if (mode == 2) idx = ((size_t)((bb * NHEAD + hh) * DHEAD + d)) * Sdim + t;
        else           idx = ((size_t)((bb * NHEAD + hh) * Tdim  + t)) * DHEAD + d;
        out[idx] = (_Float16)v;
      }
    }
  }
}

// ---------------------------------------------------------------------------
// Kernel 2: flash attention per (b,h). Block = 4 waves, each wave owns 16 rows.
// Online softmax over S in chunks of 32 columns; causal early exit.
// ---------------------------------------------------------------------------
__global__ __launch_bounds__(128)
void flash_attn(const _Float16* __restrict__ Qb, const _Float16* __restrict__ Kb,
                const _Float16* __restrict__ Vtb, const int* __restrict__ kmask,
                float* __restrict__ attnout) {
  const int bh = blockIdx.x;      // b*16 + h
  const int b  = bh >> 4;
  const int hh = bh & 15;
  const int t0 = blockIdx.y * 64;
  const int wave = threadIdx.x >> 5;
  const int lane = threadIdx.x & 31;
  const int hl   = lane >> 4;
  const int l16  = lane & 15;
  const int rowbase = t0 + wave * 16;

  const _Float16* Q  = Qb  + (size_t)bh * Tdim * DHEAD;
  const _Float16* K  = Kb  + (size_t)bh * Sdim * DHEAD;
  const _Float16* Vt = Vtb + (size_t)bh * DHEAD * Sdim;

  __shared__ __attribute__((aligned(16))) _Float16 Pt[4][16][32];

  // Q A-fragments for dh 0..31 and 32..63 — live for the whole S loop
  V16 qa0, qa1;
  {
    const _Float16* qrow = Q + (size_t)(rowbase + l16) * DHEAD;
    qa0.h[0] = *(const v8h*)(qrow + 16 * hl);
    qa0.h[1] = *(const v8h*)(qrow + 8 + 16 * hl);
    qa1.h[0] = *(const v8h*)(qrow + 32 + 16 * hl);
    qa1.h[1] = *(const v8h*)(qrow + 40 + 16 * hl);
  }

  const v8f vzero = {0.f,0.f,0.f,0.f,0.f,0.f,0.f,0.f};
  v8f o[4];
#pragma unroll
  for (int i = 0; i < 4; ++i) o[i] = vzero;
  float mrow[8], lrow[8];
#pragma unroll
  for (int r = 0; r < 8; ++r) { mrow[r] = -__builtin_inff(); lrow[r] = 0.f; }

  const float scale = 0.03125f;            // 1/sqrt(1024)
  const float NEGV  = -4294967296.0f;      // -2^32 as in reference
  const int send = t0 + 64;                // causal: no columns beyond last row

  for (int s0 = 0; s0 < send; s0 += 32) {
    // ---- scores: QK^T for 16 rows x 32 cols; load all 4 K frags, 4 WMMAs ----
    v8f sc0 = vzero, sc1 = vzero;
    {
      const _Float16* krow0 = K + (size_t)(s0 + l16) * DHEAD + 16 * hl;
      const _Float16* krow1 = K + (size_t)(s0 + 16 + l16) * DHEAD + 16 * hl;
      V16 kb00, kb01, kb10, kb11;
      kb00.h[0] = *(const v8h*)krow0;
      kb00.h[1] = *(const v8h*)(krow0 + 8);
      kb01.h[0] = *(const v8h*)(krow0 + 32);
      kb01.h[1] = *(const v8h*)(krow0 + 40);
      kb10.h[0] = *(const v8h*)krow1;
      kb10.h[1] = *(const v8h*)(krow1 + 8);
      kb11.h[0] = *(const v8h*)(krow1 + 32);
      kb11.h[1] = *(const v8h*)(krow1 + 40);
      sc0 = __builtin_amdgcn_wmma_f32_16x16x32_f16(false, qa0.v, false, kb00.v, (short)0, sc0, false, false);
      sc0 = __builtin_amdgcn_wmma_f32_16x16x32_f16(false, qa1.v, false, kb01.v, (short)0, sc0, false, false);
      sc1 = __builtin_amdgcn_wmma_f32_16x16x32_f16(false, qa0.v, false, kb10.v, (short)0, sc1, false, false);
      sc1 = __builtin_amdgcn_wmma_f32_16x16x32_f16(false, qa1.v, false, kb11.v, (short)0, sc1, false, false);
    }

    const int km0 = kmask[b * Sdim + s0 + l16];
    const int km1 = kmask[b * Sdim + s0 + 16 + l16];
    const int c0  = s0 + l16;
    const int c1  = s0 + 16 + l16;

    // ---- online softmax update + write P (f16) to LDS transposed ----
#pragma unroll
    for (int r = 0; r < 8; ++r) {
      const int row = rowbase + r + 8 * hl;
      float v0 = sc0[r] * scale;
      float v1 = sc1[r] * scale;
      if (c0 > row || km0 == 0) v0 = NEGV;
      if (c1 > row || km1 == 0) v1 = NEGV;
      float mx = fmaxf(v0, v1);
#pragma unroll
      for (int off = 1; off < 16; off <<= 1)
        mx = fmaxf(mx, __shfl_xor(mx, off, 32));
      const float mnew  = fmaxf(mrow[r], mx);
      const float alpha = __expf(mrow[r] - mnew);
      const float p0 = __expf(v0 - mnew);
      const float p1 = __expf(v1 - mnew);
      float ps = p0 + p1;
#pragma unroll
      for (int off = 1; off < 16; off <<= 1)
        ps += __shfl_xor(ps, off, 32);
      lrow[r] = lrow[r] * alpha + ps;
      mrow[r] = mnew;
#pragma unroll
      for (int dt = 0; dt < 4; ++dt) o[dt][r] *= alpha;
      Pt[wave][r + 8 * hl][l16]      = (_Float16)p0;   // C-layout -> A-layout
      Pt[wave][r + 8 * hl][16 + l16] = (_Float16)p1;   // transpose via LDS
    }
    asm volatile("s_wait_dscnt 0" ::: "memory");  // wave-internal LDS RAW fence

    // ---- P.V : A = P (16x32), B = Vt slices (4 WMMAs over dh) ----
    V16 pa;
    pa.h[0] = *(const v8h*)&Pt[wave][l16][16 * hl];
    pa.h[1] = *(const v8h*)&Pt[wave][l16][16 * hl + 8];
#pragma unroll
    for (int dt = 0; dt < 4; ++dt) {
      const _Float16* vrow = Vt + (size_t)(dt * 16 + l16) * Sdim + s0 + 16 * hl;
      V16 vb;
      vb.h[0] = *(const v8h*)vrow;
      vb.h[1] = *(const v8h*)(vrow + 8);
      o[dt] = __builtin_amdgcn_wmma_f32_16x16x32_f16(
          false, pa.v, false, vb.v, (short)0, o[dt], false, false);
    }
  }

  // ---- normalize and store f32 to (b,t,h*64+d) ----
#pragma unroll
  for (int r = 0; r < 8; ++r) {
    const int t = rowbase + r + 8 * hl;
    const float inv = 1.0f / lrow[r];
#pragma unroll
    for (int dt = 0; dt < 4; ++dt) {
      const int d = dt * 16 + l16;
      attnout[((size_t)(b * Tdim + t)) * Hdim + hh * DHEAD + d] = o[dt][r] * inv;
    }
  }
}

// ---------------------------------------------------------------------------
// Kernel 3: out = LayerNorm(attn*qmask + queries) per row (1024 features).
// ---------------------------------------------------------------------------
__global__ __launch_bounds__(256)
void mask_residual_ln(const float* __restrict__ attnout, const float* __restrict__ queries,
                      const int* __restrict__ qmask, const float* __restrict__ gamma,
                      const float* __restrict__ beta, float* __restrict__ out) {
  const int row = blockIdx.x;          // b*1024 + t
  const int tid = threadIdx.x;
  const float qm = (float)qmask[row];
  __shared__ float red[256];
  float x[4];
  float sum = 0.f;
#pragma unroll
  for (int i = 0; i < 4; ++i) {
    const int c = tid + i * 256;
    x[i] = attnout[(size_t)row * Hdim + c] * qm + queries[(size_t)row * Hdim + c];
    sum += x[i];
  }
  red[tid] = sum;
  __syncthreads();
  for (int s = 128; s > 0; s >>= 1) {
    if (tid < s) red[tid] += red[tid + s];
    __syncthreads();
  }
  const float mean = red[0] * (1.0f / (float)Hdim);
  __syncthreads();
  float vs = 0.f;
#pragma unroll
  for (int i = 0; i < 4; ++i) { const float dd = x[i] - mean; vs += dd * dd; }
  red[tid] = vs;
  __syncthreads();
  for (int s = 128; s > 0; s >>= 1) {
    if (tid < s) red[tid] += red[tid + s];
    __syncthreads();
  }
  const float rstd = rsqrtf(red[0] * (1.0f / (float)Hdim) + 1e-5f);
#pragma unroll
  for (int i = 0; i < 4; ++i) {
    const int c = tid + i * 256;
    out[(size_t)row * Hdim + c] = gamma[c] * (x[i] - mean) * rstd + beta[c];
  }
}

// ---------------------------------------------------------------------------
extern "C" void kernel_launch(void* const* d_in, const int* in_sizes, int n_in,
                              void* d_out, int out_size, void* d_ws, size_t ws_size,
                              hipStream_t stream) {
  const float* queries = (const float*)d_in[0];
  const int*   qmask   = (const int*)  d_in[1];
  const float* keys    = (const float*)d_in[2];
  const int*   kmask   = (const int*)  d_in[3];
  const float* Wq      = (const float*)d_in[4];
  const float* bq      = (const float*)d_in[5];
  const float* Wk      = (const float*)d_in[6];
  const float* bk      = (const float*)d_in[7];
  const float* Wv      = (const float*)d_in[8];
  const float* bv      = (const float*)d_in[9];
  const float* gamma   = (const float*)d_in[10];
  const float* beta    = (const float*)d_in[11];
  float* out = (float*)d_out;

  // workspace: Q f16 (8MB) | K f16 (8MB) | V^T f16 (8MB) | attnout f32 (16MB)
  char* ws = (char*)d_ws;
  _Float16* Qb  = (_Float16*)(ws);
  _Float16* Kb  = (_Float16*)(ws + ((size_t)8  << 20));
  _Float16* Vtb = (_Float16*)(ws + ((size_t)16 << 20));
  float*    att = (float*)   (ws + ((size_t)24 << 20));

  dim3 gg(32, 8);   // (4096/128) x (1024/128)
  qkv_gemm_relu<<<gg, 256, 0, stream>>>(queries, Wq, bq, Qb, 0);
  qkv_gemm_relu<<<gg, 256, 0, stream>>>(keys,    Wk, bk, Kb, 1);
  qkv_gemm_relu<<<gg, 256, 0, stream>>>(keys,    Wv, bv, Vtb, 2);
  flash_attn<<<dim3(64, 16), 128, 0, stream>>>(Qb, Kb, Vtb, kmask, att);
  mask_residual_ln<<<4096, 256, 0, stream>>>(att, queries, qmask, gamma, beta, out);
}
